// EdgeModel_13125420057114
// MI455X (gfx1250) — matrix operand order repaired and checked by reading, not compiled, not run on hardware
//
#include <hip/hip_runtime.h>

// ---------------------------------------------------------------------------
// Fused EdgeModel MLP:  out = relu(relu([src|dest|ea|u[batch]] @ W1 + b1) @ W2 + b2)
// bf16 WMMA (v_wmma_f32_16x16x32_bf16), f32 accumulate, both GEMMs fused.
//  - prep kernel converts W1^T/W2^T to bf16 in d_ws once (padded-pitch layout)
//  - persistent main kernel block-copies weights into LDS (b128 both sides),
//    then grid-strides over 64-edge tiles
//  - software-pipelined k-loop: next k-step's A/B fragments load while the
//    current 8 WMMAs issue (hides LDS latency at 1 WG/WGP occupancy)
// LDS budget: 318,464 B of the 320 KB WGP LDS.
// ---------------------------------------------------------------------------

typedef __bf16 bf16_t;
typedef __attribute__((ext_vector_type(4)))  __bf16 v4bf;
typedef __attribute__((ext_vector_type(8)))  __bf16 v8bf;
typedef __attribute__((ext_vector_type(16))) __bf16 v16bf;
typedef __attribute__((ext_vector_type(8)))  float  v8f;

#define NUM_E   320000
#define FN      128
#define FE      64
#define FU      64
#define CIN     384           // 2*FN + FE + FU
#define HID     256
#define M_TILE  64
#define PX      (CIN + 8)     // 392 : pitch (bf16 elems) for X and W1^T rows
#define PH      (HID + 8)     // 264 : pitch for H and W2^T rows
#define N_EDGE_TILES (NUM_E / M_TILE)   // 5000 exactly
#define NBLOCKS 1250
#define W1_ELEMS (HID * PX)   // 100,352 bf16
#define W2_ELEMS (FE  * PH)   //  16,896 bf16
#define WS_BYTES ((W1_ELEMS + W2_ELEMS) * 2)   // 234,496 B

// A fragment, 16x32 bf16 tile at (row, kb), pitch in elems.
// lanes 0-15 -> M=lane,    K = kb+[0..7]  then kb+[16..23]
// lanes16-31 -> M=lane-16, K = kb+[8..15] then kb+[24..31]
__device__ __forceinline__ v16bf frag_a(const bf16_t* base, int pitch,
                                        int row, int kb, int lane) {
    const int m  = row + (lane & 15);
    const int hi = lane >> 4;
    const bf16_t* p = base + m * pitch + kb + hi * 8;
    v8bf lo = *(const v8bf*)(p);
    v8bf hh = *(const v8bf*)(p + 16);
    return __builtin_shufflevector(lo, hh, 0,1,2,3,4,5,6,7,8,9,10,11,12,13,14,15);
}

// B fragment, 32x16 (K x N) stored transposed [N][K] in LDS.
// lane = N (mod 16); lanes 0-15 hold K=kb+[0..7],[16..23], lanes 16-31 K=kb+[8..15],[24..31]
__device__ __forceinline__ v16bf frag_b(const bf16_t* base, int pitch,
                                        int col, int kb, int lane) {
    const int n  = col + (lane & 15);
    const int hi = lane >> 4;
    const bf16_t* p = base + n * pitch + kb + hi * 8;
    v8bf lo = *(const v8bf*)(p);
    v8bf hh = *(const v8bf*)(p + 16);
    return __builtin_shufflevector(lo, hh, 0,1,2,3,4,5,6,7,8,9,10,11,12,13,14,15);
}

__device__ __forceinline__ v8f wmma_bf16(v16bf a, v16bf b, v8f c) {
    return __builtin_amdgcn_wmma_f32_16x16x32_bf16(
        false, a, false, b, (short)0, c, false, false);
}

// ---- one-time weight convert+transpose into d_ws (bf16, padded pitch) ----
__global__ __launch_bounds__(256) void prep_weights_kernel(
    const float* __restrict__ W1, const float* __restrict__ W2,
    bf16_t* __restrict__ wsW1, bf16_t* __restrict__ wsW2)
{
    const int stride = blockDim.x * gridDim.x;
    int tid = blockIdx.x * blockDim.x + threadIdx.x;
    for (int i = tid; i < CIN * HID; i += stride) {
        int k = i >> 8, n = i & 255;            // W1 is [K=384][N=256] row-major
        wsW1[n * PX + k] = (bf16_t)W1[i];
    }
    for (int i = tid; i < HID * FE; i += stride) {
        int k = i >> 6, n = i & 63;             // W2 is [K=256][N=64]
        wsW2[n * PH + k] = (bf16_t)W2[i];
    }
}

__global__ __launch_bounds__(256) void edge_mlp_kernel(
    const float* __restrict__ srcf, const float* __restrict__ dstf,
    const float* __restrict__ eaf,  const float* __restrict__ uf,
    const int*   __restrict__ batch,
    const float* __restrict__ W1,   const float* __restrict__ b1,
    const float* __restrict__ W2,   const float* __restrict__ b2,
    const bf16_t* __restrict__ wsW1, const bf16_t* __restrict__ wsW2,
    float* __restrict__ out)
{
    __shared__ bf16_t sW1[W1_ELEMS];     // W1^T [n][k] : 200,704 B
    __shared__ bf16_t sW2[W2_ELEMS];     // W2^T [n][k] :  33,792 B
    __shared__ bf16_t sX [M_TILE * PX];  // X tile      :  50,176 B
    __shared__ bf16_t sH [M_TILE * PH];  // H tile      :  33,792 B

    const int tid  = threadIdx.x;
    const int lane = tid & 31;
    const int wv   = tid >> 5;           // 8 waves
    const int nlo  = lane & 15;
    const int hi   = lane >> 4;

    // ---- bring bf16 weights into LDS ----
    if (wsW1 != nullptr) {
        // fast path: flat b128 copy of pre-converted, pre-transposed weights
        const v8bf* g1 = (const v8bf*)wsW1;
        v8bf*       l1 = (v8bf*)sW1;
        for (int i = tid; i < W1_ELEMS / 8; i += 256) l1[i] = g1[i];
        const v8bf* g2 = (const v8bf*)wsW2;
        v8bf*       l2 = (v8bf*)sW2;
        for (int i = tid; i < W2_ELEMS / 8; i += 256) l2[i] = g2[i];
    } else {
        // fallback: convert from f32 in-kernel
        for (int i = tid; i < CIN * HID; i += 256) {
            int k = i >> 8, n = i & 255;
            sW1[n * PX + k] = (bf16_t)W1[i];
        }
        for (int i = tid; i < HID * FE; i += 256) {
            int k = i >> 6, n = i & 63;
            sW2[n * PH + k] = (bf16_t)W2[i];
        }
    }
    __syncthreads();

    for (int t = blockIdx.x; t < N_EDGE_TILES; t += gridDim.x) {
        const int e0 = t * M_TILE;

        // prefetch next tile's activation rows (gfx1250 global_prefetch_b8)
        const int tn = t + gridDim.x;
        if (tn < N_EDGE_TILES && tid < M_TILE) {
            const int en = tn * M_TILE + tid;
            __builtin_prefetch(srcf + (size_t)en * FN, 0, 0);
            __builtin_prefetch(dstf + (size_t)en * FN, 0, 0);
            __builtin_prefetch(eaf  + (size_t)en * FE, 0, 0);
        }

        // ---- stage X tile = [src | dest | ea | u[batch]] as bf16 (float4 in) ----
        for (int i = tid; i < M_TILE * FN / 4; i += 256) {     // 2048 quads
            int r = i >> 5, q = i & 31;                         // 32 quads/row
            float4 s = ((const float4*)srcf)[(size_t)(e0 + r) * (FN / 4) + q];
            float4 d = ((const float4*)dstf)[(size_t)(e0 + r) * (FN / 4) + q];
            v4bf sb = { (bf16_t)s.x, (bf16_t)s.y, (bf16_t)s.z, (bf16_t)s.w };
            v4bf db = { (bf16_t)d.x, (bf16_t)d.y, (bf16_t)d.z, (bf16_t)d.w };
            *(v4bf*)&sX[r * PX + q * 4]      = sb;
            *(v4bf*)&sX[r * PX + FN + q * 4] = db;
        }
        for (int i = tid; i < M_TILE * FE / 4; i += 256) {     // 1024 quads
            int r = i >> 4, q = i & 15;                         // 16 quads/row
            float4 a = ((const float4*)eaf)[(size_t)(e0 + r) * (FE / 4) + q];
            int g = batch[e0 + r];
            float4 uu = ((const float4*)uf)[(size_t)g * (FU / 4) + q];
            v4bf ab = { (bf16_t)a.x,  (bf16_t)a.y,  (bf16_t)a.z,  (bf16_t)a.w };
            v4bf ub = { (bf16_t)uu.x, (bf16_t)uu.y, (bf16_t)uu.z, (bf16_t)uu.w };
            *(v4bf*)&sX[r * PX + 2 * FN + q * 4]      = ab;
            *(v4bf*)&sX[r * PX + 2 * FN + FE + q * 4] = ub;
        }
        __syncthreads();

        // ---- GEMM1: H[64x256] = relu(X[64x384] @ W1 + b1)  (waves 4x2) ----
        {
            const int mrow  = (wv >> 1) * 16;
            const int nbase = (wv & 1) * 128;
            v8f acc[8];
            #pragma unroll
            for (int n = 0; n < 8; ++n) acc[n] = (v8f)(0.f);

            v16bf a_cur = frag_a(sX, PX, mrow, 0, lane);
            v16bf b_cur[8];
            #pragma unroll
            for (int n = 0; n < 8; ++n)
                b_cur[n] = frag_b(sW1, PX, nbase + n * 16, 0, lane);

            #pragma unroll
            for (int kt = 0; kt < CIN / 32; ++kt) {
                v16bf a_nxt = a_cur;
                v16bf b_nxt[8];
                if (kt + 1 < CIN / 32) {
                    a_nxt = frag_a(sX, PX, mrow, (kt + 1) * 32, lane);
                    #pragma unroll
                    for (int n = 0; n < 8; ++n)
                        b_nxt[n] = frag_b(sW1, PX, nbase + n * 16, (kt + 1) * 32, lane);
                } else {
                    #pragma unroll
                    for (int n = 0; n < 8; ++n) b_nxt[n] = b_cur[n];
                }
                #pragma unroll
                for (int n = 0; n < 8; ++n)
                    acc[n] = wmma_bf16(a_cur, b_cur[n], acc[n]);
                a_cur = a_nxt;
                #pragma unroll
                for (int n = 0; n < 8; ++n) b_cur[n] = b_nxt[n];
            }

            #pragma unroll
            for (int n = 0; n < 8; ++n) {
                const int ncol = nbase + n * 16 + nlo;
                const float bias = b1[ncol];
                #pragma unroll
                for (int r = 0; r < 8; ++r) {
                    float v = fmaxf(acc[n][r] + bias, 0.f);
                    sH[(mrow + r + hi * 8) * PH + ncol] = (bf16_t)v;
                }
            }
        }
        __syncthreads();

        // ---- GEMM2: OUT[64x64] = relu(H[64x256] @ W2 + b2)  (waves 4x2x2) ----
        {
            const int mrow = (wv >> 1) * 16;
            const int nb2  = (wv & 1) * 32;
            v8f acc2[2];
            acc2[0] = (v8f)(0.f);
            acc2[1] = (v8f)(0.f);

            v16bf a_cur = frag_a(sH, PH, mrow, 0, lane);
            v16bf b_cur[2];
            #pragma unroll
            for (int j = 0; j < 2; ++j)
                b_cur[j] = frag_b(sW2, PH, nb2 + j * 16, 0, lane);

            #pragma unroll
            for (int kt = 0; kt < HID / 32; ++kt) {
                v16bf a_nxt = a_cur;
                v16bf b_nxt[2];
                if (kt + 1 < HID / 32) {
                    a_nxt = frag_a(sH, PH, mrow, (kt + 1) * 32, lane);
                    #pragma unroll
                    for (int j = 0; j < 2; ++j)
                        b_nxt[j] = frag_b(sW2, PH, nb2 + j * 16, (kt + 1) * 32, lane);
                } else {
                    #pragma unroll
                    for (int j = 0; j < 2; ++j) b_nxt[j] = b_cur[j];
                }
                #pragma unroll
                for (int j = 0; j < 2; ++j)
                    acc2[j] = wmma_bf16(a_cur, b_cur[j], acc2[j]);
                a_cur = a_nxt;
                #pragma unroll
                for (int j = 0; j < 2; ++j) b_cur[j] = b_nxt[j];
            }

            #pragma unroll
            for (int j = 0; j < 2; ++j) {
                const int ncol = nb2 + j * 16 + nlo;
                const float bias = b2[ncol];
                #pragma unroll
                for (int r = 0; r < 8; ++r) {
                    float v = fmaxf(acc2[j][r] + bias, 0.f);
                    const int e = e0 + mrow + r + hi * 8;
                    out[(size_t)e * FE + ncol] = v;
                }
            }
        }
        __syncthreads();   // protect sX/sH before next tile overwrites them
    }
}

extern "C" void kernel_launch(void* const* d_in, const int* in_sizes, int n_in,
                              void* d_out, int out_size, void* d_ws, size_t ws_size,
                              hipStream_t stream) {
    (void)in_sizes; (void)n_in; (void)out_size;
    const float* srcf  = (const float*)d_in[0];
    const float* dstf  = (const float*)d_in[1];
    const float* eaf   = (const float*)d_in[2];
    const float* uf    = (const float*)d_in[3];
    const int*   batch = (const int*)  d_in[4];
    const float* W1    = (const float*)d_in[5];
    const float* b1    = (const float*)d_in[6];
    const float* W2    = (const float*)d_in[7];
    const float* b2    = (const float*)d_in[8];
    float* out = (float*)d_out;

    bf16_t* wsW1 = nullptr;
    bf16_t* wsW2 = nullptr;
    if (d_ws != nullptr && ws_size >= (size_t)WS_BYTES) {
        wsW1 = (bf16_t*)d_ws;
        wsW2 = wsW1 + W1_ELEMS;
        prep_weights_kernel<<<64, 256, 0, stream>>>(W1, W2, wsW1, wsW2);
    }
    edge_mlp_kernel<<<NBLOCKS, 256, 0, stream>>>(srcf, dstf, eaf, uf, batch,
                                                 W1, b1, W2, b2, wsW1, wsW2, out);
}
